// GatedLinearAttention_84774064488855
// MI455X (gfx1250) — matrix operand order repaired and checked by reading, not compiled, not run on hardware
//
#include <hip/hip_runtime.h>
#include <hip/hip_bf16.h>
#include <math.h>
#include <stdint.h>

// ---------------- problem constants ----------------
static constexpr int Bb  = 2;
static constexpr int Sq  = 4096;
static constexpr int Dm  = 1024;
static constexpr int Hh  = 4;
static constexpr int DKc = 128;
static constexpr int DVc = 256;
static constexpr int KDc = Hh * DKc;   // 512
static constexpr int VDc = Hh * DVc;   // 1024
static constexpr int Cc  = 64;         // chunk length
static constexpr int NCc = Sq / Cc;    // 64 chunks
static constexpr int Mrows = Bb * Sq;  // 8192

typedef __attribute__((ext_vector_type(16))) _Float16 v16h;
typedef __attribute__((ext_vector_type(8)))  _Float16 v8h;
typedef __attribute__((ext_vector_type(8)))  float    v8f;

// ---------------- WMMA helpers (wave32 fragment layouts per CDNA5 ISA) ----------------
__device__ __forceinline__ v8f zero8() {
  v8f z;
#pragma unroll
  for (int e = 0; e < 8; ++e) z[e] = 0.0f;
  return z;
}

__device__ __forceinline__ v8f wmma_f16(v16h a, v16h b, v8f c) {
  return __builtin_amdgcn_wmma_f32_16x16x32_f16(false, a, false, b, (short)0, c, false, false);
}

// A fragment: 16x32 f16 tile, row-major, leading dim ld (halves); two aligned 16-B runs.
__device__ __forceinline__ v16h load_a_frag(const _Float16* p, int ld) {
  int lane = threadIdx.x & 31;
  int row  = lane & 15;
  int kb   = (lane >> 4) * 8;
  const v8h* p0 = reinterpret_cast<const v8h*>(p + row * ld + kb);
  const v8h* p1 = reinterpret_cast<const v8h*>(p + row * ld + kb + 16);
  v8h lo = *p0, hi = *p1;
  return __builtin_shufflevector(lo, hi, 0, 1, 2, 3, 4, 5, 6, 7, 8, 9, 10, 11, 12, 13, 14, 15);
}

// B fragment from a TRANSPOSED tile (tile[n][k]): two aligned 16-B runs per lane (same k order as A).
__device__ __forceinline__ v16h load_bt_frag(const _Float16* p, int ldk) {
  int lane = threadIdx.x & 31;
  const _Float16* q = p + (lane & 15) * ldk + (lane >> 4) * 8;
  const v8h* p0 = reinterpret_cast<const v8h*>(q);
  const v8h* p1 = reinterpret_cast<const v8h*>(q + 16);
  v8h lo = *p0, hi = *p1;
  return __builtin_shufflevector(lo, hi, 0, 1, 2, 3, 4, 5, 6, 7, 8, 9, 10, 11, 12, 13, 14, 15);
}

// Low 32 bits of a generic pointer to LDS = group-segment byte address (ISA 10.2).
__device__ __forceinline__ unsigned lds_addr32(const void* p) {
  return (unsigned)(uintptr_t)p;
}

// ---------------- Kernel 1: tiled WMMA GEMM, C = alpha * (A @ B) ----------------
// A: MxK f32 row-major, B: KxN f32 row-major, C: MxN f32. N,K compile-time so the
// strided B-column loads fold into immediate-offset clauses.
// 256 threads (8 waves). BM=64, BN=128, BK=32. B staged TRANSPOSED in LDS.
template <int N, int K>
__global__ __launch_bounds__(256)
void wmma_gemm(const float* __restrict__ A, const float* __restrict__ B,
               float* __restrict__ Cm, float alpha) {
  constexpr int BM = 64, BN = 128, BK = 32;
  constexpr int LDA = 40, LDBT = 40;            // 80-byte strides: aligned + conflict-free
  __shared__ _Float16 As[BM * LDA];             // [m][k]
  __shared__ _Float16 BsT[BN * LDBT];           // [n][k]  (transposed)

  const int tid  = threadIdx.x;
  const int wave = tid >> 5;
  const int lane = tid & 31;
  const int m0   = blockIdx.y * BM;
  const int n0   = blockIdx.x * BN;
  const int mt   = (wave & 3) * 16;
  const int nb   = (wave >> 2) * 64;

  v8f acc[4];
#pragma unroll
  for (int j = 0; j < 4; ++j) acc[j] = zero8();

  // B staging: one column per thread, 16 consecutive k each (coalesced across lanes)
  const int bn  = tid & 127;
  const int bkh = (tid >> 7) * 16;

  for (int k0 = 0; k0 < K; k0 += BK) {
    // stage A tile 64x32 (f32 -> f16), contiguous b128 path
    {
      int r = tid >> 2;
      int c = (tid & 3) * 8;
      const float4* src = reinterpret_cast<const float4*>(&A[(size_t)(m0 + r) * K + k0 + c]);
      float4 f0 = src[0], f1 = src[1];
      _Float16* dst = &As[r * LDA + c];
      dst[0] = (_Float16)f0.x; dst[1] = (_Float16)f0.y; dst[2] = (_Float16)f0.z; dst[3] = (_Float16)f0.w;
      dst[4] = (_Float16)f1.x; dst[5] = (_Float16)f1.y; dst[6] = (_Float16)f1.z; dst[7] = (_Float16)f1.w;
    }
    // stage B tile 32x128 transposed: 16 immediate-offset b32 loads, two ds_store_b128
    {
      const float* src = &B[(size_t)(k0 + bkh) * N + n0 + bn];
      float tmp[16];
#pragma unroll
      for (int i = 0; i < 16; ++i) tmp[i] = src[i * N];     // constant offsets
      v8h lo, hi;
#pragma unroll
      for (int e = 0; e < 8; ++e) { lo[e] = (_Float16)tmp[e]; hi[e] = (_Float16)tmp[e + 8]; }
      v8h* dst = reinterpret_cast<v8h*>(&BsT[bn * LDBT + bkh]);
      dst[0] = lo;
      dst[1] = hi;
    }
    // prefetch next K tile while this one is consumed
    if (k0 + BK < K) {
      __builtin_prefetch(&A[(size_t)(m0 + (tid >> 2)) * K + k0 + BK + (tid & 3) * 8], 0, 3);
      __builtin_prefetch(&B[(size_t)(k0 + BK + bkh) * N + n0 + bn], 0, 3);
    }
    __syncthreads();
    v16h a = load_a_frag(&As[mt * LDA], LDA);
#pragma unroll
    for (int j = 0; j < 4; ++j) {
      v16h b = load_bt_frag(&BsT[(nb + j * 16) * LDBT], LDBT);
      acc[j] = wmma_f16(a, b, acc[j]);
    }
    __syncthreads();
  }

  const int col   = lane & 15;
  const int rbase = (lane >> 4) * 8;
#pragma unroll
  for (int j = 0; j < 4; ++j)
#pragma unroll
    for (int r = 0; r < 8; ++r)
      Cm[(size_t)(m0 + mt + rbase + r) * N + n0 + nb + j * 16 + col] = acc[j][r] * alpha;
}

// ---------------- Kernel 2: t = x @ Wgk1  (N = RANK = 16) ----------------
__global__ __launch_bounds__(256)
void lowrank_t(const float* __restrict__ x, const float* __restrict__ Wgk1,
               float* __restrict__ t, int M, int K) {
  int gid = blockIdx.x * blockDim.x + threadIdx.x;
  int m = gid >> 4, r = gid & 15;
  const float* xr = x + (size_t)m * K;
  float s = 0.0f;
  for (int k = 0; k < K; k += 4) {
    s += xr[k]     * Wgk1[(k)     * 16 + r];
    s += xr[k + 1] * Wgk1[(k + 1) * 16 + r];
    s += xr[k + 2] * Wgk1[(k + 2) * 16 + r];
    s += xr[k + 3] * Wgk1[(k + 3) * 16 + r];
  }
  t[gid] = s;
}

// ---------------- Kernel 3: gk = log_sigmoid(t @ Wgk2 + bgk) / 16 ----------------
__global__ __launch_bounds__(256)
void gk_kernel(const float* __restrict__ t, const float* __restrict__ Wgk2,
               const float* __restrict__ bgk, float* __restrict__ gk) {
  int gid = blockIdx.x * blockDim.x + threadIdx.x;
  int m = gid >> 9, j = gid & 511;
  float s = bgk[j];
  const float* tr = t + (size_t)m * 16;
#pragma unroll
  for (int r = 0; r < 16; ++r) s += tr[r] * Wgk2[r * 512 + j];
  float ls = fminf(s, 0.0f) - log1pf(expf(-fabsf(s)));
  gk[gid] = ls * (1.0f / 16.0f);
}

// ---------------- Kernel 4: chunked GLA scan ----------------
static constexpr int LDQ = 136;  // [t][d] tiles: 64 x 128 (+8 pad)
static constexpr int LDV = 72;   // 64-wide tiles (+8 pad)
static constexpr int LDS_T = 136;// transposed state [col][d]: 128-wide (+8 pad)

__global__ __launch_bounds__(256)
void gla_kernel(const float* __restrict__ q, const float* __restrict__ k,
                const float* __restrict__ v, const float* __restrict__ gk,
                float* __restrict__ o) {
  extern __shared__ char smem_raw[];
  float*    s_gc   = (float*)smem_raw;                        // [64][128] gk -> cumsum  32768 B
  float*    s_st   = (float*)(smem_raw + 32768);              // [128][64] state f32     32768 B
  float*    s_gl   = (float*)(smem_raw + 65536);              // [128] g_last              512 B
  float*    s_egl  = (float*)(smem_raw + 66048);              // [128] exp(g_last)         512 B
  _Float16* s_qt   = (_Float16*)(smem_raw + 66560);           // [64][LDQ]  q*e^g        17408 B
  _Float16* s_ki   = s_qt  + 64 * LDQ;                        // [64][LDQ]  k*e^-g       17408 B
  _Float16* s_kdT  = s_ki  + 64 * LDQ;                        // [128][LDV] k_dec^T      18432 B
  _Float16* s_vT   = s_kdT + 128 * LDV;                       // [64][LDV]  v^T           9216 B
  _Float16* s_A    = s_vT  + 64 * LDV;                        // [64][LDV]  attn          9216 B
  _Float16* s_sthT = s_A   + 64 * LDV;                        // [64][LDS_T] state^T     17408 B

  const int tid   = threadIdx.x;
  const int wave  = tid >> 5;
  const int lane  = tid & 31;
  const int col16 = lane & 15;
  const int rbase = (lane >> 4) * 8;

  const int bid = blockIdx.x;          // B*H*4
  const int vb  = bid & 3;
  const int bh  = bid >> 2;
  const int h   = bh & (Hh - 1);
  const int b   = bh >> 2;
  const size_t row0 = (size_t)b * Sq;

  for (int i = tid; i < 128 * 64; i += 256) s_st[i] = 0.0f;
  for (int i = tid; i < 64 * LDS_T; i += 256) s_sthT[i] = (_Float16)0.0f;
  __syncthreads();

  const unsigned gc_lds = lds_addr32(s_gc);

  for (int nc = 0; nc < NCc; ++nc) {
    const int t0 = nc * Cc;

    // -- async DMA of the gk tile straight into LDS (ASYNCcnt path):
    //    each wave copies one 512-B row (32 lanes x 16 B); 8 issues cover 64 rows.
    {
      const float* gsrc = gk + (row0 + t0) * KDc + (size_t)h * DKc;
#pragma unroll
      for (int it = 0; it < 8; ++it) {
        int trow = it * 8 + wave;
        uint64_t ga = (uint64_t)(uintptr_t)(gsrc + (size_t)trow * KDc) + (unsigned)(lane * 16);
        unsigned  la = gc_lds + (unsigned)(trow * 512 + lane * 16);
        asm volatile("global_load_async_to_lds_b128 %0, %1, off"
                     :: "v"(la), "v"(ga) : "memory");
      }
      asm volatile("s_wait_asynccnt 0x0" ::: "memory");
    }
    __syncthreads();

    // -- per-channel cumsum over the chunk
    if (tid < 128) {
      float gsum = 0.0f;
      for (int t = 0; t < 64; ++t) { gsum += s_gc[t * 128 + tid]; s_gc[t * 128 + tid] = gsum; }
      s_gl[tid]  = gsum;
      s_egl[tid] = expf(gsum);
    }
    __syncthreads();

    // -- build q_t, k_inv, k_dec^T tiles ; stage v transposed
    for (int i = tid; i < 64 * 128; i += 256) {
      int t = i >> 7, d = i & 127;
      float gc = s_gc[i];
      size_t gidx = (row0 + t0 + t) * KDc + h * DKc + d;
      float qf = q[gidx], kf = k[gidx];
      s_qt[t * LDQ + d]  = (_Float16)(qf * expf(gc));
      s_ki[t * LDQ + d]  = (_Float16)(kf * expf(-gc));
      s_kdT[d * LDV + t] = (_Float16)(kf * expf(s_gl[d] - gc));
    }
    for (int i = tid; i < 64 * 64; i += 256) {
      int t = i >> 6, c = i & 63;
      s_vT[c * LDV + t] = (_Float16)(v[(row0 + t0 + t) * VDc + h * DVc + vb * 64 + c]);
    }
    __syncthreads();

    // -- A = q_t @ k_inv^T (64x64, K=128), tril mask, -> f16
#pragma unroll
    for (int i = 0; i < 2; ++i) {
      int ti = wave * 2 + i;
      int mt = (ti >> 2) * 16, nt = (ti & 3) * 16;
      v8f acc = zero8();
#pragma unroll
      for (int kk = 0; kk < 4; ++kk) {
        v16h a  = load_a_frag(&s_qt[mt * LDQ + kk * 32], LDQ);
        v16h bb = load_bt_frag(&s_ki[nt * LDQ + kk * 32], LDQ);
        acc = wmma_f16(a, bb, acc);
      }
#pragma unroll
      for (int r = 0; r < 8; ++r) {
        int rr = mt + rbase + r, cc = nt + col16;
        s_A[rr * LDV + cc] = (rr >= cc) ? (_Float16)acc[r] : (_Float16)0.0f;
      }
    }
    __syncthreads();

    // -- o = A @ v + q_t @ state (64x64)
#pragma unroll
    for (int i = 0; i < 2; ++i) {
      int ti = wave * 2 + i;
      int mt = (ti >> 2) * 16, nt = (ti & 3) * 16;
      v8f acc = zero8();
#pragma unroll
      for (int kk = 0; kk < 2; ++kk) {           // intra: K = 64 (time)
        v16h a  = load_a_frag(&s_A[mt * LDV + kk * 32], LDV);
        v16h bb = load_bt_frag(&s_vT[nt * LDV + kk * 32], LDV);
        acc = wmma_f16(a, bb, acc);
      }
#pragma unroll
      for (int kk = 0; kk < 4; ++kk) {           // inter: K = 128 (dk)
        v16h a  = load_a_frag(&s_qt[mt * LDQ + kk * 32], LDQ);
        v16h bb = load_bt_frag(&s_sthT[nt * LDS_T + kk * 32], LDS_T);
        acc = wmma_f16(a, bb, acc);
      }
#pragma unroll
      for (int r = 0; r < 8; ++r) {
        int rr = t0 + mt + rbase + r;
        o[(row0 + rr) * VDc + h * DVc + vb * 64 + nt + col16] = acc[r];
      }
    }
    __syncthreads();

    // -- kv = k_dec^T @ v ; state = e^{g_last} * state + kv
#pragma unroll
    for (int i = 0; i < 4; ++i) {
      int ti = wave * 4 + i;
      int dt = (ti >> 2) * 16, nt = (ti & 3) * 16;
      v8f acc = zero8();
#pragma unroll
      for (int kk = 0; kk < 2; ++kk) {           // K = 64 (time)
        v16h a  = load_a_frag(&s_kdT[dt * LDV + kk * 32], LDV);
        v16h bb = load_bt_frag(&s_vT[nt * LDV + kk * 32], LDV);
        acc = wmma_f16(a, bb, acc);
      }
#pragma unroll
      for (int r = 0; r < 8; ++r) {
        int d = dt + rbase + r;
        int cc = nt + col16;
        float ns = s_st[d * 64 + cc] * s_egl[d] + acc[r];
        s_st[d * 64 + cc] = ns;
        s_sthT[cc * LDS_T + d] = (_Float16)ns;
      }
    }
    __syncthreads();
  }
}

// ---------------- Kernel 5: RMSNorm * norm_w * SiLU(g) ----------------
__global__ __launch_bounds__(256)
void norm_silu(const float* __restrict__ o, const float* __restrict__ g,
               const float* __restrict__ norm_w, float* __restrict__ on) {
  __shared__ float red[256];
  int bsh = blockIdx.x;                 // (b*S+s)*H + h
  int row = bsh >> 2, h = bsh & 3;
  int j   = threadIdx.x;
  size_t idx = (size_t)row * VDc + h * DVc + j;
  float val = o[idx];
  red[j] = val * val;
  __syncthreads();
  for (int s = 128; s > 0; s >>= 1) {
    if (j < s) red[j] += red[j + s];
    __syncthreads();
  }
  float rs = rsqrtf(red[0] * (1.0f / 256.0f) + 1e-5f);
  float gv = g[idx];
  float silu = gv / (1.0f + expf(-gv));
  on[idx] = val * rs * norm_w[j] * silu;
}

// ---------------- launcher ----------------
extern "C" void kernel_launch(void* const* d_in, const int* in_sizes, int n_in,
                              void* d_out, int out_size, void* d_ws, size_t ws_size,
                              hipStream_t stream) {
  (void)in_sizes; (void)n_in; (void)out_size; (void)ws_size;
  const float* x    = (const float*)d_in[0];
  const float* Wq   = (const float*)d_in[1];
  const float* Wk   = (const float*)d_in[2];
  const float* Wv   = (const float*)d_in[3];
  const float* Wgk1 = (const float*)d_in[4];
  const float* Wgk2 = (const float*)d_in[5];
  const float* bgk  = (const float*)d_in[6];
  const float* Wg   = (const float*)d_in[7];
  const float* Wo   = (const float*)d_in[8];
  const float* nw   = (const float*)d_in[9];
  float* out = (float*)d_out;

  float* ws  = (float*)d_ws;
  float* qb  = ws; ws += (size_t)Mrows * KDc;
  float* kb  = ws; ws += (size_t)Mrows * KDc;
  float* vbf = ws; ws += (size_t)Mrows * VDc;
  float* gb  = ws; ws += (size_t)Mrows * VDc;
  float* gkb = ws; ws += (size_t)Mrows * KDc;
  float* tb  = ws; ws += (size_t)Mrows * 16;
  float* ob  = ws; ws += (size_t)Mrows * VDc;
  float* onb = ws; ws += (size_t)Mrows * VDc;

  const dim3 blk(256);
  const float qscale = 0.08838834764831845f;   // DK^-0.5

  wmma_gemm<KDc, Dm><<<dim3(KDc / 128, Mrows / 64), blk, 0, stream>>>(x, Wq, qb,  qscale);
  wmma_gemm<KDc, Dm><<<dim3(KDc / 128, Mrows / 64), blk, 0, stream>>>(x, Wk, kb,  1.0f);
  wmma_gemm<VDc, Dm><<<dim3(VDc / 128, Mrows / 64), blk, 0, stream>>>(x, Wv, vbf, 1.0f);
  wmma_gemm<VDc, Dm><<<dim3(VDc / 128, Mrows / 64), blk, 0, stream>>>(x, Wg, gb,  1.0f);
  lowrank_t<<<dim3((Mrows * 16) / 256), blk, 0, stream>>>(x, Wgk1, tb, Mrows, Dm);
  gk_kernel<<<dim3((Mrows * KDc) / 256), blk, 0, stream>>>(tb, Wgk2, bgk, gkb);

  // dynamic LDS: 2x32768 + 2x512 + 17408*2 + 18432 + 9216*2 + 17408 = 155648 B
  size_t smem = 155648;
  gla_kernel<<<dim3(Bb * Hh * (DVc / 64)), blk, smem, stream>>>(qb, kb, vbf, gkb, ob);

  norm_silu<<<dim3(Mrows * Hh), blk, 0, stream>>>(ob, gb, nw, onb);
  wmma_gemm<Dm, Dm><<<dim3(Dm / 128, Mrows / 64), blk, 0, stream>>>(onb, Wo, out, 1.0f);
}
